// JBF_block_29643864277435
// MI455X (gfx1250) — matrix operand order repaired and checked by reading, not compiled, not run on hardware
//
#include <hip/hip_runtime.h>

// ---- CDNA5 WMMA / TDM types ----
typedef __attribute__((ext_vector_type(16))) _Float16 v16h;
typedef __attribute__((ext_vector_type(8)))  float    v8f;
typedef __attribute__((ext_vector_type(4)))  unsigned u32x4;
typedef __attribute__((ext_vector_type(8)))  int      i32x8;
typedef __attribute__((ext_vector_type(4)))  int      i32x4;

#if defined(__HIP_DEVICE_COMPILE__) && __has_builtin(__builtin_amdgcn_tensor_load_to_lds)
#define USE_TDM 1
#else
#define USE_TDM 0
#endif

#define DIMD 26
#define DIMH 122
#define DIMW 122
#define V_OUT (DIMD*DIMH*DIMW)      // 386984 output voxels
#define GSTR  360                   // LDS row stride (halves) for g: conflict-free (180 dw, step 52 mod 64)
#define RKSTR 132                   // LDS row stride (halves) for rk (66 dw, step 2 mod 64)
#define NW    2                     // waves per block (wave32), 16 voxels per wave

// workspace layout (bytes)
#define WS_DK  0                    // 27 f32
#define WS_W0  512                  // 8 ntiles * 11 ksteps * 32 lanes * 16 halves = 45056 h = 90112 B
#define WS_W1  (512 + 90112)        // 2 ntiles * 4 ksteps * 32 lanes * 16 halves = 4096 h = 8192 B

__device__ __forceinline__ void dec_v(int v, int& d, int& h, int& w) {
    d = v / (DIMH * DIMW);
    int r = v - d * (DIMH * DIMW);
    h = r / DIMW;
    w = r - h * DIMW;
}

#if USE_TDM
// LDS byte offset of a generic pointer to a __shared__ object.
__device__ __forceinline__ unsigned lds_off(const void* p) {
    return (unsigned)(unsigned long long)
        (__attribute__((address_space(3))) const void*)p;
}

// Issue one TDM copy: 8192 B (W1 fragments) global -> LDS.
// D# per CDNA5 ISA:
//  group0: count=1 | lds_addr | global_addr[56:0] | type=2
//  group1: data_size=3 (8B) | tensor_dim0=1024 | tensor_dim1=1 | tile_dim0=1024
//          tile_dim1=1 | tile_dim2=0 | tensor_dim0_stride=1024
__device__ __forceinline__ void tdm_load_w1(const _Float16* gsrc, unsigned ldsByte) {
    unsigned long long ga = (unsigned long long)gsrc;
    u32x4 g0 = { 1u,                                     // count=1 (valid user D#)
                 ldsByte,                                // lds_addr
                 (unsigned)(ga & 0xffffffffull),         // global_addr lo
                 (unsigned)((ga >> 32) & 0x1ffffffull)   // global_addr hi (25 bits)
                 | 0x80000000u };                        // type=2 ("image")
    i32x8 g1 = { (int)(3u << 16),   // wg_mask=0, data_size=3 (8B units)
                 (int)0x04000000,   // tensor_dim0=1024 (low16 << 16)
                 (int)0x00010000,   // tensor_dim0 hi=0 | tensor_dim1 lo16=1
                 (int)0x04000000,   // tensor_dim1 hi=0 | tile_dim0=1024 (<<16)
                 1,                 // tile_dim1=1, tile_dim2=0
                 1024,              // tensor_dim0_stride lo32
                 0, 0 };            // stride hi / tensor_dim1_stride = 0
    i32x4 gz = { 0, 0, 0, 0 };
#if __clang_major__ >= 23
    i32x8 gz8 = { 0, 0, 0, 0, 0, 0, 0, 0 };
    __builtin_amdgcn_tensor_load_to_lds(g0, g1, gz, gz, gz8, 0);
#else
    __builtin_amdgcn_tensor_load_to_lds(g0, g1, gz, gz, 0);
#endif
}
#endif // USE_TDM

// ---------------------------------------------------------------------------
// Kernel A: precompute dk (27 f32) and W0/W1 in WMMA B-fragment layout (f16).
// B-fragment layout for 16x16x32 f16: lane gives N = lane&15; half h gives
// K = 16*(lane>>4) + h  (contiguous order chosen here; A uses the same).
// ---------------------------------------------------------------------------
__global__ void jbf_precompute(const float* __restrict__ dn9,
                               const float* __restrict__ rw0, const float* __restrict__ rb0,
                               const float* __restrict__ rw1, const float* __restrict__ rb1,
                               const float* __restrict__ dw0, const float* __restrict__ db0,
                               const float* __restrict__ dw1, const float* __restrict__ db1,
                               float* __restrict__ dkOut,
                               _Float16* __restrict__ W0f, _Float16* __restrict__ W1f) {
    __shared__ float dk0s[125];
    const int tid = threadIdx.x;

    if (tid < 125) {
        int od = tid / 25, rr = tid - od * 25, oh = rr / 5, ow = rr - oh * 5;
        float acc = db0[0];
        #pragma unroll
        for (int kd = 0; kd < 3; ++kd)
            #pragma unroll
            for (int kh = 0; kh < 3; ++kh)
                #pragma unroll
                for (int kw = 0; kw < 3; ++kw)
                    acc += dw0[kd*9 + kh*3 + kw] *
                           dn9[(od+kd+1)*81 + (oh+kh+1)*9 + (ow+kw+1)];
        dk0s[tid] = fmaxf(acc, 0.f);
    }
    __syncthreads();
    if (tid < 27) {
        int od = tid / 9, rr = tid - od * 9, oh = rr / 3, ow = rr - oh * 3;
        float acc = db1[0];
        #pragma unroll
        for (int kd = 0; kd < 3; ++kd)
            #pragma unroll
            for (int kh = 0; kh < 3; ++kh)
                #pragma unroll
                for (int kw = 0; kw < 3; ++kw)
                    acc += dw1[kd*9 + kh*3 + kw] *
                           dk0s[(od+kd)*25 + (oh+kh)*5 + (ow+kw)];
        dkOut[tid] = fmaxf(acc, 0.f);
    }

    // W0: 343x125 (K x N), zero-padded to 352x128, baked into B fragments.
    for (int idx = tid; idx < 8 * 11 * 32 * 16; idx += blockDim.x) {
        int hh   = idx & 15;
        int lane = (idx >> 4) & 31;
        int rest = idx >> 9;
        int s = rest % 11;
        int t = rest / 11;
        int n = t * 16 + (lane & 15);
        int k = s * 32 + (lane >> 4) * 16 + hh;
        float val = 0.f;
        if (k < 343 && n < 125) {
            int pd = k / 49, pr = k - pd * 49, ph = pr / 7, pw = pr - ph * 7;
            int od = n / 25, nr = n - od * 25, oh = nr / 5, ow = nr - oh * 5;
            int kd = pd - od, kh = ph - oh, kw = pw - ow;
            if ((unsigned)kd < 3u && (unsigned)kh < 3u && (unsigned)kw < 3u)
                val = rw0[kd*9 + kh*3 + kw];
        }
        W0f[idx] = (_Float16)val;
    }

    // W1: 125x27 (K x N), padded to 128x32.
    for (int idx = tid; idx < 2 * 4 * 32 * 16; idx += blockDim.x) {
        int hh   = idx & 15;
        int lane = (idx >> 4) & 31;
        int rest = idx >> 9;
        int s = rest % 4;
        int t = rest / 4;
        int n = t * 16 + (lane & 15);
        int k = s * 32 + (lane >> 4) * 16 + hh;
        float val = 0.f;
        if (k < 125 && n < 27) {
            int pd = k / 25, pr = k - pd * 25, ph = pr / 5, pw = pr - ph * 5;
            int od = n / 9,  nr = n - od * 9,  oh = nr / 3, ow = nr - oh * 3;
            int kd = pd - od, kh = ph - oh, kw = pw - ow;
            if ((unsigned)kd < 3u && (unsigned)kh < 3u && (unsigned)kw < 3u)
                val = rw1[kd*9 + kh*3 + kw];
        }
        W1f[idx] = (_Float16)val;
    }
}

// ---------------------------------------------------------------------------
// Kernel B: main JBF. One wave (32 lanes) handles 16 output voxels (M=16).
// conv1 = G(16x343) @ W0(343x125): 11 K-steps x 8 N-tiles = 88 WMMAs/wave.
// conv2 = RK(16x125) @ W1(125x27):  4 K-steps x 2 N-tiles =  8 WMMAs/wave.
// A-fragment K mapping: K = 16*(lane>>4) + h, same contiguous order as the B fragments.
// D layout: N = lane&15, M = vgpr + 8*(lane>>4).
// ---------------------------------------------------------------------------
__global__ __launch_bounds__(NW * 32)
void jbf_main(const float* __restrict__ x, const float* __restrict__ guide,
              const float* __restrict__ rb0p, const float* __restrict__ rb1p,
              const float* __restrict__ dkG,
              const _Float16* __restrict__ W0f, const _Float16* __restrict__ W1f,
              float* __restrict__ out) {
    __shared__ __align__(16) _Float16 gS[NW][16][GSTR];
    __shared__ __align__(16) _Float16 rkS[NW][16][RKSTR];
    __shared__ float wS[NW][16][32];
    __shared__ float cenS[NW][16];
    __shared__ float dkS[32];
#if USE_TDM
    __shared__ __align__(16) _Float16 w1S[4096];   // TDM destination (8 KB)
#endif

    const int tid  = threadIdx.x;
    const int wv   = tid >> 5;
    const int lane = tid & 31;
    const int grp  = lane >> 4;
    const int m    = lane & 15;
    const int base = (blockIdx.x * NW + wv) * 16;

    const float rb0v = rb0p[0];
    const float rb1v = rb1p[0];

    // ---- stage 0: TDM for W1 frags, dk to LDS, guide centers to LDS ----
#if USE_TDM
    // Each wave issues the same (idempotent) DMA and later waits its own
    // TENSORcnt -> no cross-wave completion-visibility problem.
    tdm_load_w1(W1f, lds_off(&w1S[0]));
#endif
    if (tid < 27) dkS[tid] = dkG[tid];
    if (lane < 16) {
        int v = base + lane;
        float c = 0.f;
        if (v < V_OUT) {
            int d, h, w; dec_v(v, d, h, w);
            c = guide[(d + 3) * 16384 + (h + 3) * 128 + (w + 3)];
            __builtin_prefetch(&x[(d + 2) * 16384 + (h + 2) * 128 + (w + 2)], 0, 0);
        }
        cenS[wv][lane] = c;
    }
    __syncthreads();

    // ---- stage 1: g = |patch - center| into LDS (f16), zero-pad K ----
    for (int i = lane; i < 16 * 343; i += 32) {
        int mm = i / 343;
        int k  = i - mm * 343;
        int v  = base + mm;
        float val = 0.f;
        if (v < V_OUT) {
            int d, h, w; dec_v(v, d, h, w);
            int pd = k / 49, pr = k - pd * 49, ph = pr / 7, pw = pr - ph * 7;
            val = fabsf(guide[(d + pd) * 16384 + (h + ph) * 128 + (w + pw)] - cenS[wv][mm]);
        }
        gS[wv][mm][k] = (_Float16)val;
    }
    for (int i = lane; i < 16 * (GSTR - 343); i += 32) {
        int mm = i / (GSTR - 343);
        int k  = 343 + (i - mm * (GSTR - 343));
        gS[wv][mm][k] = (_Float16)0.f;
    }
    __syncthreads();

    // ---- stage 2: conv1 GEMM via v_wmma_f32_16x16x32_f16 ----
    v8f zero8 = {0.f, 0.f, 0.f, 0.f, 0.f, 0.f, 0.f, 0.f};
    v8f acc[8];
    #pragma unroll
    for (int t = 0; t < 8; ++t) acc[t] = zero8;

    const char* grow = (const char*)&gS[wv][m][0];
    #pragma unroll
    for (int s = 0; s < 11; ++s) {
        union { v16h v; unsigned u[8]; } A;
        #pragma unroll
        for (int q = 0; q < 2; ++q)
            #pragma unroll
            for (int j = 0; j < 4; ++j) {
                int K = s * 32 + grp * 16 + q * 8 + 2 * j;   // even -> b32 LDS load
                A.u[q * 4 + j] = *(const unsigned*)(grow + 2 * K);
            }
        // batch all 8 B-fragments of this K-step: one load clause, one wait,
        // then 8 back-to-back WMMAs instead of load->wait->wmma per tile.
        u32x4 bq[16];
        #pragma unroll
        for (int t = 0; t < 8; ++t) {
            const u32x4* bp = (const u32x4*)(W0f + ((size_t)((t * 11 + s) * 32 + lane) * 16));
            bq[2 * t]     = bp[0];
            bq[2 * t + 1] = bp[1];
        }
        #pragma unroll
        for (int t = 0; t < 8; ++t) {
            union { v16h v; u32x4 q[2]; } B;
            B.q[0] = bq[2 * t];
            B.q[1] = bq[2 * t + 1];
            acc[t] = __builtin_amdgcn_wmma_f32_16x16x32_f16(
                false, A.v, false, B.v, (short)0, acc[t], false, false);
        }
    }

    // bias + relu -> rkS (f16), zero the N=125..127 pad
    #pragma unroll
    for (int t = 0; t < 8; ++t)
        #pragma unroll
        for (int r = 0; r < 8; ++r) {
            int n  = t * 16 + m;
            int mm = r + grp * 8;
            float val = fmaxf(acc[t][r] + rb0v, 0.f);
            rkS[wv][mm][n] = (n < 125) ? (_Float16)val : (_Float16)0.f;
        }
    __syncthreads();

    // ---- stage 3: conv2 GEMM (B fragments from TDM-staged LDS if available) ----
#if USE_TDM
#if __has_builtin(__builtin_amdgcn_s_wait_tensorcnt)
    __builtin_amdgcn_s_wait_tensorcnt(0);
#else
    asm volatile("s_wait_tensorcnt 0x0" ::: "memory");
#endif
    asm volatile("" ::: "memory");
    const char* w1base = (const char*)w1S;
#endif

    v8f a2[2];
    a2[0] = zero8; a2[1] = zero8;
    const char* rrow = (const char*)&rkS[wv][m][0];
    #pragma unroll
    for (int s = 0; s < 4; ++s) {
        union { v16h v; unsigned u[8]; } A;
        #pragma unroll
        for (int q = 0; q < 2; ++q)
            #pragma unroll
            for (int j = 0; j < 4; ++j) {
                int K = s * 32 + grp * 16 + q * 8 + 2 * j;
                A.u[q * 4 + j] = *(const unsigned*)(rrow + 2 * K);
            }
        #pragma unroll
        for (int t = 0; t < 2; ++t) {
            union { v16h v; u32x4 q[2]; } B;
#if USE_TDM
            const u32x4* bp = (const u32x4*)(w1base + ((size_t)((t * 4 + s) * 32 + lane) * 32));
#else
            const u32x4* bp = (const u32x4*)(W1f + ((size_t)((t * 4 + s) * 32 + lane) * 16));
#endif
            B.q[0] = bp[0];
            B.q[1] = bp[1];
            a2[t] = __builtin_amdgcn_wmma_f32_16x16x32_f16(
                false, A.v, false, B.v, (short)0, a2[t], false, false);
        }
    }

    // weights = dk * relu(conv2 + rb1) + 1e-10 -> wS
    #pragma unroll
    for (int t = 0; t < 2; ++t)
        #pragma unroll
        for (int r = 0; r < 8; ++r) {
            int n3 = t * 16 + m;
            int mm = r + grp * 8;
            if (n3 < 27) {
                float val = fmaxf(a2[t][r] + rb1v, 0.f);
                wS[wv][mm][n3] = dkS[n3] * val + 1e-10f;
            }
        }
    __syncthreads();

    // ---- stage 4: weighted average of 3^3 x-patch, store ----
    if (lane < 16) {
        int v = base + lane;
        if (v < V_OUT) {
            int d, h, w; dec_v(v, d, h, w);
            float num = 0.f, den = 0.f;
            #pragma unroll
            for (int o3 = 0; o3 < 27; ++o3) {
                int kd = o3 / 9, rr = o3 - kd * 9, kh = rr / 3, kw = rr - kh * 3;
                float wt = wS[wv][lane][o3];
                float xv = x[(d + 2 + kd) * 16384 + (h + 2 + kh) * 128 + (w + 2 + kw)];
                num += xv * wt;
                den += wt;
            }
            out[v] = num / den;
        }
    }
}

// ---------------------------------------------------------------------------
extern "C" void kernel_launch(void* const* d_in, const int* in_sizes, int n_in,
                              void* d_out, int out_size, void* d_ws, size_t ws_size,
                              hipStream_t stream) {
    const float* x     = (const float*)d_in[0];
    const float* dn9   = (const float*)d_in[1];
    const float* guide = (const float*)d_in[2];
    const float* rw0   = (const float*)d_in[3];
    const float* rb0   = (const float*)d_in[4];
    const float* rw1   = (const float*)d_in[5];
    const float* rb1   = (const float*)d_in[6];
    const float* dw0   = (const float*)d_in[7];
    const float* db0   = (const float*)d_in[8];
    const float* dw1   = (const float*)d_in[9];
    const float* db1   = (const float*)d_in[10];

    float*    dkW = (float*)((char*)d_ws + WS_DK);
    _Float16* W0f = (_Float16*)((char*)d_ws + WS_W0);
    _Float16* W1f = (_Float16*)((char*)d_ws + WS_W1);

    jbf_precompute<<<1, 256, 0, stream>>>(dn9, rw0, rb0, rw1, rb1,
                                          dw0, db0, dw1, db1, dkW, W0f, W1f);

    const int voxPerBlock = NW * 16;
    const int nb = (V_OUT + voxPerBlock - 1) / voxPerBlock;
    jbf_main<<<nb, NW * 32, 0, stream>>>(x, guide, rb0, rb1, dkW, W0f, W1f,
                                         (float*)d_out);
}
